// DiscreteFullyConnectedQFunction_53618371723592
// MI455X (gfx1250) — matrix operand order repaired and checked by reading, not compiled, not run on hardware
//
#include <hip/hip_runtime.h>
#include <hip/hip_bf16.h>

// ---------- types ----------
typedef __bf16 bf16_t;
typedef __attribute__((ext_vector_type(16))) __bf16        v16bf;
typedef __attribute__((ext_vector_type(8)))  float         v8f;
typedef __attribute__((ext_vector_type(4)))  unsigned int  u32x4;
typedef __attribute__((ext_vector_type(8)))  int           i32x8;
typedef __attribute__((ext_vector_type(4)))  float         f32x4;

union ABFrag { u32x4 q[2]; v16bf v; };

// f32 -> bf16 round-to-nearest-even (pure integer ops: always legal)
static __device__ __forceinline__ unsigned short f2bf_u(float f) {
  unsigned u = __builtin_bit_cast(unsigned, f);
  u = (u + 0x7fffu + ((u >> 16) & 1u)) >> 16;
  return (unsigned short)u;
}
static __device__ __forceinline__ bf16_t f2bf(float f) {
  unsigned short s = f2bf_u(f);
  return __builtin_bit_cast(bf16_t, s);
}
static __device__ __forceinline__ float bf2f(bf16_t h) {
  unsigned short s = __builtin_bit_cast(unsigned short, h);
  unsigned u = ((unsigned)s) << 16;
  return __builtin_bit_cast(float, u);
}
static __device__ __forceinline__ unsigned pack2bf(float lo, float hi) {
  return (unsigned)f2bf_u(lo) | ((unsigned)f2bf_u(hi) << 16);
}

// ---------- Tensor Data Mover: 2D tile load global->LDS (ISA cdna5 ch.8) ----
// D# group0: count=1, lds_addr, global_addr(57b), type=2 ("image")
// D# group1: data_size=2B, pad cfg, tensor_dim0/1, tile_dim0/1, dim0 stride
static __device__ __forceinline__ void tdm_load_2d(unsigned lds_off,
                                                   unsigned long long gaddr,
                                                   unsigned tile0, unsigned tile1,
                                                   unsigned td0, unsigned td1,
                                                   unsigned long long stride0,
                                                   unsigned pad_en, unsigned pad_int,
                                                   unsigned pad_amt) {
  u32x4 g0;
  g0.x = 1u;                                                   // count=1
  g0.y = lds_off;                                              // lds_addr
  g0.z = (unsigned)gaddr;                                      // global_addr[31:0]
  g0.w = (unsigned)((gaddr >> 32) & 0x01FFFFFFull) | (2u << 30); // addr[56:32] | type=2
  i32x8 g1;
  g1[0] = (int)((1u << 16) | (pad_en << 20) | (pad_int << 22) | (pad_amt << 25)); // data_size=2B
  g1[1] = (int)((td0 & 0xffffu) << 16);                        // tensor_dim0 lo16 @ [63:48]
  g1[2] = (int)(((td0 >> 16) & 0xffffu) | ((td1 & 0xffffu) << 16));
  g1[3] = (int)(((td1 >> 16) & 0xffffu) | ((tile0 & 0xffffu) << 16));
  g1[4] = (int)(tile1 & 0xffffu);                              // tile_dim1 (tile_dim2=0)
  g1[5] = (int)(unsigned)(stride0 & 0xffffffffull);            // dim0 stride lo32
  g1[6] = (int)(unsigned)((stride0 >> 32) & 0xffffull);        // stride hi16 (dim1 stride=0)
  g1[7] = 0;
  asm volatile("tensor_load_to_lds %0, %1" :: "s"(g0), "s"(g1) : "memory");
}

// ---------- problem constants ----------
#define BB   64
#define NN   1000
#define NJ   50
#define HH   128
#define HC   256
#define ROWS (BB * NN)        // 64000
#define OBSR (BB * NJ)        // 3200

// =====================================================================
// Layer-1 neighbor pooling: pooled1[b,n,d] = sum_m adj[b,n,m]*feat[b,m,d]
// =====================================================================
__global__ void k_pooled1(const float* __restrict__ adj,
                          const float* __restrict__ feat,
                          float* __restrict__ out) {
  int lane = threadIdx.x & 31, wave = threadIdx.x >> 5;
  int row = blockIdx.x * 8 + wave;                 // < 64000
  int b = row / NN, n = row % NN;
  const float* ap = adj + (size_t)b * NN * NN + (size_t)n * NN;
  const float* fp = feat + (size_t)b * NN * 2;
  float a0 = 0.f, a1 = 0.f;
  for (int m = lane; m < NN; m += 32) {
    float a = ap[m];
    a0 += a * fp[2 * m];
    a1 += a * fp[2 * m + 1];
  }
  for (int o = 16; o > 0; o >>= 1) {
    a0 += __shfl_down(a0, o, 32);
    a1 += __shfl_down(a1, o, 32);
  }
  if (lane == 0) { out[(size_t)row * 2] = a0; out[(size_t)row * 2 + 1] = a1; }
}

// =====================================================================
// y = pooled1 @ W1(2x128) + b1, fused BN stats
// =====================================================================
__global__ void k_lin2(const float* __restrict__ pooled,
                       const float* __restrict__ W1,
                       const float* __restrict__ b1,
                       float* __restrict__ y,
                       float* __restrict__ stats) {
  int f = threadIdx.x;
  float w0 = W1[f], w1 = W1[HH + f], bb = b1[f];
  float s = 0.f, sq = 0.f;
  int base = blockIdx.x * 256;
  for (int r = 0; r < 256; ++r) {
    int row = base + r;
    float p0 = pooled[(size_t)row * 2], p1 = pooled[(size_t)row * 2 + 1];
    float v = p0 * w0 + p1 * w1 + bb;
    y[(size_t)row * HH + f] = v;
    s += v; sq += v * v;
  }
  atomicAdd(&stats[f], s);
  atomicAdd(&stats[HH + f], sq);
}

// =====================================================================
// BN (training stats) + ReLU, elementwise, F=128
// =====================================================================
__global__ void k_bn_relu(const float* __restrict__ X,
                          const float* __restrict__ stats,
                          const float* __restrict__ g,
                          const float* __restrict__ bt,
                          float invcnt,
                          float* __restrict__ outF,
                          bf16_t* __restrict__ outB,
                          unsigned long long total) {
  unsigned long long i = (unsigned long long)blockIdx.x * blockDim.x + threadIdx.x;
  if (i >= total) return;
  int f = (int)(i & (HH - 1));
  float mu = stats[f] * invcnt;
  float var = stats[HH + f] * invcnt - mu * mu;
  float yv = g[f] * (X[i] - mu) * rsqrtf(var + 1e-5f) + bt[f];
  yv = yv > 0.f ? yv : 0.f;
  if (outF) outF[i] = yv;
  if (outB) outB[i] = f2bf(yv);
}

// =====================================================================
// Pack f32 weight [K,N] into WMMA B-fragment order (bf16)
// =====================================================================
__global__ void k_packW(const float* __restrict__ W, bf16_t* __restrict__ dst, int N) {
  int lane = threadIdx.x;                       // 32
  int ntiles = N >> 4;
  int nt = blockIdx.x % ntiles;
  int ks = blockIdx.x / ntiles;
  int ncol = nt * 16 + (lane & 15);
  int kbase = ks * 32 + (lane >> 4) * 16;
  bf16_t* d = dst + ((size_t)blockIdx.x * 32 + lane) * 16;
#pragma unroll
  for (int j = 0; j < 16; ++j) d[j] = f2bf(W[(size_t)(kbase + j) * N + ncol]);
}

// Pack h1 (bf16 row-major) into per-batch B fragments, K zero-padded to 1024
__global__ void k_packH(const bf16_t* __restrict__ h, bf16_t* __restrict__ hp) {
  unsigned idx = blockIdx.x * blockDim.x + threadIdx.x;   // < 524288
  int lane = idx & 31; unsigned t = idx >> 5;
  int nt = t & 7;  t >>= 3;
  int ks = t & 31; t >>= 5;
  int b = (int)t;                                          // < 64
  int ncol = nt * 16 + (lane & 15);
  int kbase = ks * 32 + (lane >> 4) * 16;
  bf16_t* d = hp + (size_t)b * 131072 + (((size_t)(ks * 8 + nt)) * 32 + lane) * 16;
  bf16_t z = f2bf(0.f);
#pragma unroll
  for (int j = 0; j < 16; ++j) {
    int m = kbase + j;
    d[j] = (m < NN) ? h[((size_t)b * NN + m) * HH + ncol] : z;
  }
}

// =====================================================================
// WMMA row-GEMM:  C[rows,N] = A_bf16[rows,K] x Bpack + bias
//   mode 0: store f32 + BN stats atomics ; mode 1: ReLU -> bf16
// B panel hoisted into registers (KSTEPS fragments per wave).
// A tile (16 x K bf16) DMA'd to LDS by the Tensor Data Mover (wave 0),
// with TDM row padding (4 dwords) for bank-staggered LDS rows.
// =====================================================================
template <int KSTEPS>
__global__ __launch_bounds__(256) void k_rowgemm(const bf16_t* __restrict__ A,
                                                 const bf16_t* __restrict__ Bp,
                                                 const float* __restrict__ bias,
                                                 float* __restrict__ outF,
                                                 bf16_t* __restrict__ outB,
                                                 float* __restrict__ stats,
                                                 int rows, int N, int rt, int mode) {
  constexpr int K = KSTEPS * 32;
  constexpr int SROW = K + 8;                 // LDS row stride (elements), 16B multiple
  constexpr unsigned PADI = (KSTEPS == 4) ? 5u : 6u;   // 2^(p+1) dwords = K/2
  __shared__ bf16_t sA[16 * SROW];

  int lane = threadIdx.x & 31, wave = threadIdx.x >> 5;
  int nt = wave + blockIdx.y * 8;
  int ln = lane & 15, half = lane >> 4;
  int col = nt * 16 + ln;
  int ntiles = N >> 4;
  int rtiles = rows >> 4;
  float ssum = 0.f, ssq = 0.f;
  float bcol = bias[col];

  // hoist B panel into registers
  ABFrag bfr[KSTEPS];
#pragma unroll
  for (int ks = 0; ks < KSTEPS; ++ks) {
    const u32x4* bp = (const u32x4*)(Bp + ((size_t)(ks * ntiles + nt) * 32 + lane) * 16);
    bfr[ks].q[0] = bp[0]; bfr[ks].q[1] = bp[1];
  }

  unsigned lds_off = (unsigned)(size_t)(&sA[0]);

  for (int i = 0; i < rt; ++i) {
    int t = blockIdx.x * rt + i;
    if (t >= rtiles) break;
    int rowbase = t << 4;

    __syncthreads();                                    // all waves done reading sA
    if (wave == 0) {
      unsigned long long ga = (unsigned long long)(size_t)(A + (size_t)rowbase * K);
      // tile 16 rows x K cols of bf16, row stride K; pad 4 dwords per K/2-dword row
      tdm_load_2d(lds_off, ga, (unsigned)K, 16u, (unsigned)K, 1u << 20,
                  (unsigned long long)K, 1u, PADI, 3u);
      __builtin_amdgcn_s_wait_tensorcnt(0);
    }
    __syncthreads();                                    // sA visible to all waves

    v8f c = {0.f, 0.f, 0.f, 0.f, 0.f, 0.f, 0.f, 0.f};
#pragma unroll
    for (int ks = 0; ks < KSTEPS; ++ks) {
      int kk = ks << 5;
      ABFrag ua;
      const u32x4* p1 = (const u32x4*)(&sA[ln * SROW + kk + 16 * half]);
      const u32x4* p2 = (const u32x4*)(&sA[ln * SROW + kk + 16 * half + 8]);
      ua.q[0] = p1[0]; ua.q[1] = p2[0];
      c = __builtin_amdgcn_wmma_f32_16x16x32_bf16(false, ua.v, false, bfr[ks].v,
                                                  (short)0, c, false, false);
    }
#pragma unroll
    for (int e = 0; e < 8; ++e) {
      int r = rowbase + e + 8 * half;
      float v = c[e] + bcol;
      if (mode == 0) {
        outF[(size_t)r * N + col] = v;
        ssum += v; ssq += v * v;
      } else {
        v = v > 0.f ? v : 0.f;
        outB[(size_t)r * N + col] = f2bf(v);
      }
    }
  }
  if (mode == 0) {
    atomicAdd(&stats[col], ssum);
    atomicAdd(&stats[N + col], ssq);
  }
}

// =====================================================================
// Dominant GEMM: per batch b, pooled2[1000,128] = bf16(adj[b]) x hpack[b]
// 64 rows x 128 cols per block: adj read from HBM exactly once.
// A tiles staged+converted cooperatively into LDS in fragment layout
// (lane stride 24 elements keeps ds_load at 2-way minimum).
// =====================================================================
__global__ __launch_bounds__(256) void k_adj_gemm(const float* __restrict__ adj,
                                                  const bf16_t* __restrict__ hp,
                                                  bf16_t* __restrict__ out) {
  constexpr int LSTR = 24;                    // padded lane stride (elements)
  __shared__ bf16_t sA[4 * 32 * LSTR];        // 4 M-tiles x 32 lanes x 16(+8 pad)

  int tid = threadIdx.x;
  int lane = tid & 31, wave = tid >> 5;
  int b = blockIdx.y;
  int nt = wave;
  int ln = lane & 15, half = lane >> 4;
  int col = nt * 16 + ln;
  int row0 = blockIdx.x << 6;                 // 64 rows per block

  // staging role: thread -> (source row, 8-wide k chunk)
  int sr = tid >> 2, cc = tid & 3;
  int grow = row0 + sr;
  const float* ap = adj + (size_t)b * NN * NN + (size_t)grow * NN;
  int laneS = (sr & 15) + 16 * (cc >> 1);     // fragment lane
  bf16_t* sdst = &sA[(size_t)((sr >> 4) * 32 + laneS) * LSTR + 8 * (cc & 1)];

  const bf16_t* hb = hp + (size_t)b * 131072;

  const v8f vz = {0.f, 0.f, 0.f, 0.f, 0.f, 0.f, 0.f, 0.f};
  v8f c[4] = {vz, vz, vz, vz};

  for (int ks = 0; ks < 32; ++ks) {
    int kk = ks << 5;
    __syncthreads();                          // previous tile fully consumed
    // ---- cooperative stage + f32->bf16 convert (once per block) ----
    float v[8];
    int gk = kk + 8 * cc;
    if (grow < NN && (kk + 32) <= NN) {
      f32x4 x0 = *(const f32x4*)(ap + gk);
      f32x4 x1 = *(const f32x4*)(ap + gk + 4);
#pragma unroll
      for (int j = 0; j < 4; ++j) { v[j] = x0[j]; v[4 + j] = x1[j]; }
    } else {
#pragma unroll
      for (int j = 0; j < 8; ++j) {
        int k = gk + j;
        v[j] = (grow < NN && k < NN) ? ap[k] : 0.f;
      }
    }
    u32x4 pk;
#pragma unroll
    for (int p = 0; p < 4; ++p) pk[p] = pack2bf(v[2 * p], v[2 * p + 1]);
    *(u32x4*)sdst = pk;
    __syncthreads();
    // ---- compute: one B fragment, 4 M-tiles ----
    ABFrag ub;
    const u32x4* bp = (const u32x4*)(hb + ((size_t)(ks * 8 + nt) * 32 + lane) * 16);
    ub.q[0] = bp[0]; ub.q[1] = bp[1];
#pragma unroll
    for (int t4 = 0; t4 < 4; ++t4) {
      ABFrag ua;
      const u32x4* pa = (const u32x4*)(&sA[(size_t)(t4 * 32 + lane) * LSTR]);
      ua.q[0] = pa[0]; ua.q[1] = pa[1];
      c[t4] = __builtin_amdgcn_wmma_f32_16x16x32_bf16(false, ua.v, false, ub.v,
                                                      (short)0, c[t4], false, false);
    }
  }
#pragma unroll
  for (int t4 = 0; t4 < 4; ++t4) {
#pragma unroll
    for (int e = 0; e < 8; ++e) {
      int r = row0 + t4 * 16 + e + 8 * half;
      if (r < NN) out[((size_t)b * NN + r) * HH + col] = f2bf(c[t4][e]);
    }
  }
}

// graph pooling: h_pooled[b,f] = sum_n gp[b,n]*h2[b,n,f]
__global__ void k_graphpool(const float* __restrict__ gp,
                            const float* __restrict__ h2,
                            float* __restrict__ hpool,
                            bf16_t* __restrict__ hpoolb) {
  int b = blockIdx.x, f = threadIdx.x;            // 64 x 128
  const float* g = gp + (size_t)b * NN;
  const float* hb = h2 + (size_t)b * NN * HH;
  float acc = 0.f;
  for (int n = 0; n < NN; ++n) acc += g[n] * hb[(size_t)n * HH + f];
  hpool[b * HH + f] = acc;
  hpoolb[b * HH + f] = f2bf(acc);
}

// obs[b,j,:] = [ h2[b,cand[b,j],:128] , h_pooled[b,:128] ]  -> bf16
__global__ void k_obs(const float* __restrict__ h2,
                      const float* __restrict__ hpool,
                      const int* __restrict__ cand,
                      bf16_t* __restrict__ obs) {
  int idx = blockIdx.x * 256 + threadIdx.x;       // < 819200
  int f = idx & 255;
  int t = idx >> 8;
  int j = t % NJ, b = t / NJ;
  float v;
  if (f < HH) v = h2[((size_t)b * NN + cand[b * NJ + j]) * HH + f];
  else        v = hpool[b * HH + (f - HH)];
  obs[idx] = f2bf(v);
}

// final head linear K->1: wave per row
__global__ void k_headdot(const bf16_t* __restrict__ act,
                          const float* __restrict__ W,
                          const float* __restrict__ b,
                          float* __restrict__ out, int K) {
  int lane = threadIdx.x & 31, wave = threadIdx.x >> 5;
  int row = blockIdx.x * 8 + wave;
  const bf16_t* a = act + (size_t)row * K;
  float acc = 0.f;
  for (int k = lane; k < K; k += 32) acc += bf2f(a[k]) * W[k];
  for (int o = 16; o > 0; o >>= 1) acc += __shfl_down(acc, o, 32);
  if (lane == 0) out[row] = acc + b[0];
}

// dueling combine + action gather
__global__ void k_combine(const float* __restrict__ q,
                          const float* __restrict__ v,
                          const int* __restrict__ mask,
                          const int* __restrict__ actions,
                          float* __restrict__ out) {
  int b = threadIdx.x;                            // 64
  const float* qb = q + b * NJ;
  float s = 0.f, cnt = 0.f;
  for (int j = 0; j < NJ; ++j) {
    float m = mask[b * NJ + j] ? 1.f : 0.f;
    s += qb[j] * m; cnt += m;
  }
  float mm = s / cnt;
  int a = actions[b];
  out[b] = v[b] + qb[a] - mm;
}

// =====================================================================
extern "C" void kernel_launch(void* const* d_in, const int* in_sizes, int n_in,
                              void* d_out, int out_size, void* d_ws, size_t ws_size,
                              hipStream_t stream) {
  const float* adj   = (const float*)d_in[0];
  const float* feat  = (const float*)d_in[1];
  const int*   cand  = (const int*)d_in[2];
  const float* gpool = (const float*)d_in[3];
  const int*   acts  = (const int*)d_in[4];
  const int*   amask = (const int*)d_in[5];
  const float* l1W1 = (const float*)d_in[6];  const float* l1b1 = (const float*)d_in[7];
  const float* l1g1 = (const float*)d_in[8];  const float* l1B1 = (const float*)d_in[9];
  const float* l1W2 = (const float*)d_in[10]; const float* l1b2 = (const float*)d_in[11];
  const float* l1g  = (const float*)d_in[12]; const float* l1B  = (const float*)d_in[13];
  const float* l2W1 = (const float*)d_in[14]; const float* l2b1 = (const float*)d_in[15];
  const float* l2g1 = (const float*)d_in[16]; const float* l2B1 = (const float*)d_in[17];
  const float* l2W2 = (const float*)d_in[18]; const float* l2b2 = (const float*)d_in[19];
  const float* l2g  = (const float*)d_in[20]; const float* l2B  = (const float*)d_in[21];
  const float* cW1 = (const float*)d_in[22]; const float* cb1 = (const float*)d_in[23];
  const float* cW2 = (const float*)d_in[24]; const float* cb2 = (const float*)d_in[25];
  const float* cW3 = (const float*)d_in[26]; const float* cb3 = (const float*)d_in[27];
  const float* vW1 = (const float*)d_in[28]; const float* vb1 = (const float*)d_in[29];
  const float* vW2 = (const float*)d_in[30]; const float* vb2 = (const float*)d_in[31];
  const float* vW3 = (const float*)d_in[32]; const float* vb3 = (const float*)d_in[33];

  char* ws = (char*)d_ws;
  size_t off = 0;
  auto take = [&](size_t bytes) -> char* {
    char* p = ws + off;
    off = (off + bytes + 255) & ~(size_t)255;
    return p;
  };
  float*  stats   = (float*)take(1024 * 4);
  float*  pooled1 = (float*)take((size_t)ROWS * 2 * 4);
  float*  bufA    = (float*)take((size_t)ROWS * HH * 4);
  bf16_t* bfbuf   = (bf16_t*)take((size_t)ROWS * HH * 2);
  bf16_t* hpack   = (bf16_t*)take((size_t)BB * 131072 * 2);
  bf16_t* pW2L1   = (bf16_t*)take((size_t)HH * HH * 2);
  bf16_t* pW1L2   = (bf16_t*)take((size_t)HH * HH * 2);
  bf16_t* pW2L2   = (bf16_t*)take((size_t)HH * HH * 2);
  bf16_t* pCW1    = (bf16_t*)take((size_t)HC * HC * 2);
  bf16_t* pCW2    = (bf16_t*)take((size_t)HC * HC * 2);
  bf16_t* pVW1    = (bf16_t*)take((size_t)HH * HC * 2);
  bf16_t* pVW2    = (bf16_t*)take((size_t)HC * HC * 2);
  bf16_t* obs     = (bf16_t*)take((size_t)OBSR * HC * 2);
  bf16_t* act1    = (bf16_t*)take((size_t)OBSR * HC * 2);
  bf16_t* act2    = (bf16_t*)take((size_t)OBSR * HC * 2);
  float*  qbuf    = (float*)take((size_t)OBSR * 4);
  float*  hpool   = (float*)take((size_t)BB * HH * 4);
  bf16_t* hpoolb  = (bf16_t*)take((size_t)BB * HH * 2);
  bf16_t* vact1   = (bf16_t*)take((size_t)BB * HC * 2);
  bf16_t* vact2   = (bf16_t*)take((size_t)BB * HC * 2);
  float*  vbuf    = (float*)take((size_t)BB * 4);
  if (off > ws_size) return;

  float* out = (float*)d_out;
  const float invcnt = 1.0f / (float)ROWS;

  hipMemsetAsync(stats, 0, 1024 * 4, stream);

  k_packW<<<dim3(4 * 8), 32, 0, stream>>>(l1W2, pW2L1, HH);
  k_packW<<<dim3(4 * 8), 32, 0, stream>>>(l2W1, pW1L2, HH);
  k_packW<<<dim3(4 * 8), 32, 0, stream>>>(l2W2, pW2L2, HH);
  k_packW<<<dim3(8 * 16), 32, 0, stream>>>(cW1, pCW1, HC);
  k_packW<<<dim3(8 * 16), 32, 0, stream>>>(cW2, pCW2, HC);
  k_packW<<<dim3(4 * 16), 32, 0, stream>>>(vW1, pVW1, HC);
  k_packW<<<dim3(8 * 16), 32, 0, stream>>>(vW2, pVW2, HC);

  // ---- GIN layer 1 ----
  k_pooled1<<<dim3(ROWS / 8), 256, 0, stream>>>(adj, feat, pooled1);
  k_lin2<<<dim3(ROWS / 256), 128, 0, stream>>>(pooled1, l1W1, l1b1, bufA, stats + 0);
  k_bn_relu<<<dim3(32000), 256, 0, stream>>>(bufA, stats + 0, l1g1, l1B1, invcnt,
                                             nullptr, bfbuf, (unsigned long long)ROWS * HH);
  k_rowgemm<4><<<dim3(250, 1), 256, 0, stream>>>(bfbuf, pW2L1, l1b2, bufA, nullptr,
                                                 stats + 256, ROWS, HH, 16, 0);
  k_bn_relu<<<dim3(32000), 256, 0, stream>>>(bufA, stats + 256, l1g, l1B, invcnt,
                                             nullptr, bfbuf, (unsigned long long)ROWS * HH);

  // ---- GIN layer 2 ----
  k_packH<<<dim3(2048), 256, 0, stream>>>(bfbuf, hpack);
  k_adj_gemm<<<dim3(16, 64), 256, 0, stream>>>(adj, hpack, bfbuf);   // pooled2 (bf16)
  k_rowgemm<4><<<dim3(250, 1), 256, 0, stream>>>(bfbuf, pW1L2, l2b1, bufA, nullptr,
                                                 stats + 512, ROWS, HH, 16, 0);
  k_bn_relu<<<dim3(32000), 256, 0, stream>>>(bufA, stats + 512, l2g1, l2B1, invcnt,
                                             nullptr, bfbuf, (unsigned long long)ROWS * HH);
  k_rowgemm<4><<<dim3(250, 1), 256, 0, stream>>>(bfbuf, pW2L2, l2b2, bufA, nullptr,
                                                 stats + 768, ROWS, HH, 16, 0);
  k_bn_relu<<<dim3(32000), 256, 0, stream>>>(bufA, stats + 768, l2g, l2B, invcnt,
                                             bufA, nullptr, (unsigned long long)ROWS * HH);

  // ---- pooling / gather ----
  k_graphpool<<<dim3(BB), 128, 0, stream>>>(gpool, bufA, hpool, hpoolb);
  k_obs<<<dim3(3200), 256, 0, stream>>>(bufA, hpool, cand, obs);

  // ---- critic head (3200 rows: 256->256->256->1) ----
  k_rowgemm<8><<<dim3(50, 2), 256, 0, stream>>>(obs, pCW1, cb1, nullptr, act1,
                                                nullptr, OBSR, HC, 4, 1);
  k_rowgemm<8><<<dim3(50, 2), 256, 0, stream>>>(act1, pCW2, cb2, nullptr, act2,
                                                nullptr, OBSR, HC, 4, 1);
  k_headdot<<<dim3(OBSR / 8), 256, 0, stream>>>(act2, cW3, cb3, qbuf, HC);

  // ---- value head (64 rows: 128->256->256->1) ----
  k_rowgemm<4><<<dim3(1, 2), 256, 0, stream>>>(hpoolb, pVW1, vb1, nullptr, vact1,
                                               nullptr, BB, HC, 4, 1);
  k_rowgemm<8><<<dim3(1, 2), 256, 0, stream>>>(vact1, pVW2, vb2, nullptr, vact2,
                                               nullptr, BB, HC, 4, 1);
  k_headdot<<<dim3(BB / 8), 256, 0, stream>>>(vact2, vW3, vb3, vbuf, HC);

  // ---- dueling combine ----
  k_combine<<<dim3(1), 64, 0, stream>>>(qbuf, vbuf, amask, acts, out);
}
